// APPNPNet_27084063769017
// MI455X (gfx1250) — matrix operand order repaired and checked by reading, not compiled, not run on hardware
//
#include <hip/hip_runtime.h>
#include <hip/hip_bf16.h>

typedef __attribute__((ext_vector_type(16))) __bf16 v16bf;
typedef __attribute__((ext_vector_type(8)))  __bf16 v8bf;
typedef __attribute__((ext_vector_type(8)))  float  v8f;
typedef __attribute__((ext_vector_type(4)))  float  v4f;

#define IN_C  512
#define HID_C 256
#define OUT_C 64

// ---------------- weight conversion f32 -> bf16 ----------------
__global__ void cvt_f32_bf16_kernel(const float* __restrict__ in,
                                    __bf16* __restrict__ out, int n) {
  int i = blockIdx.x * 256 + threadIdx.x;
  if (i < n) out[i] = (__bf16)in[i];
}

// ---------------- GEMM1: h1 = relu(x @ W1^T + b1), bf16 out ----------------
// wave tile: 16 rows x 64 cols. 8 waves/block.
__global__ __launch_bounds__(256) void gemm1_kernel(
    const float* __restrict__ x, const __bf16* __restrict__ W1,
    const float* __restrict__ b1, __bf16* __restrict__ h1, int n_nodes) {
  const int tid  = threadIdx.x;
  const int lane = tid & 31;
  const int wv   = tid >> 5;
  const int wgl  = blockIdx.x * 8 + wv;
  const int mTiles = (n_nodes + 15) >> 4;
  const int mTile  = wgl >> 2;
  const int nGroup = wgl & 3;
  if (mTile >= mTiles) return;           // wave-uniform: EXEC stays all-ones

  const int m16  = lane & 15;
  const int half = lane >> 4;
  int row = mTile * 16 + m16;
  if (row >= n_nodes) row = n_nodes - 1;
  const float* xrow = x + (size_t)row * IN_C;

  v8f acc[4] = {v8f{}, v8f{}, v8f{}, v8f{}};

  for (int k0 = 0; k0 < IN_C; k0 += 32) {
    const int ka = k0 + half * 16;       // A element i<8  -> K = ka+i
    const int kb = k0 + half * 16 + 8;   // A element i>=8 -> K = kb+(i-8)
    v4f a0 = *(const v4f*)(xrow + ka);
    v4f a1 = *(const v4f*)(xrow + ka + 4);
    v4f a2 = *(const v4f*)(xrow + kb);
    v4f a3 = *(const v4f*)(xrow + kb + 4);
    v16bf a;
#pragma unroll
    for (int i = 0; i < 4; ++i) {
      a[i]      = (__bf16)a0[i];
      a[i + 4]  = (__bf16)a1[i];
      a[i + 8]  = (__bf16)a2[i];
      a[i + 12] = (__bf16)a3[i];
    }
    const int kcol = k0 + half * 16;     // B element i -> K = kcol+i (contiguous)
#pragma unroll
    for (int j = 0; j < 4; ++j) {
      const int n = nGroup * 64 + j * 16 + m16;
      v16bf b = *(const v16bf*)(W1 + (size_t)n * IN_C + kcol);
      acc[j] = __builtin_amdgcn_wmma_f32_16x16x32_bf16(
          false, a, false, b, (short)0, acc[j], false, false);
    }
  }

#pragma unroll
  for (int j = 0; j < 4; ++j) {
    const int col  = nGroup * 64 + j * 16 + m16;
    const float bs = b1[col];
#pragma unroll
    for (int v = 0; v < 8; ++v) {
      const int r = mTile * 16 + v + 8 * half;
      if (r < n_nodes) {
        float val = acc[j][v] + bs;
        val = val > 0.f ? val : 0.f;
        h1[(size_t)r * HID_C + col] = (__bf16)val;
      }
    }
  }
}

// ---------------- GEMM2: h0 = h1 @ W2^T + b2, f32 out ----------------
__global__ __launch_bounds__(256) void gemm2_kernel(
    const __bf16* __restrict__ h1, const __bf16* __restrict__ W2,
    const float* __restrict__ b2, float* __restrict__ h0, int n_nodes) {
  const int tid  = threadIdx.x;
  const int lane = tid & 31;
  const int wv   = tid >> 5;
  const int mTiles = (n_nodes + 15) >> 4;
  const int mTile  = blockIdx.x * 8 + wv;
  if (mTile >= mTiles) return;           // wave-uniform

  const int m16  = lane & 15;
  const int half = lane >> 4;
  int row = mTile * 16 + m16;
  if (row >= n_nodes) row = n_nodes - 1;
  const __bf16* arow = h1 + (size_t)row * HID_C;

  v8f acc[4] = {v8f{}, v8f{}, v8f{}, v8f{}};

  for (int k0 = 0; k0 < HID_C; k0 += 32) {
    v8bf alo = *(const v8bf*)(arow + k0 + half * 16);
    v8bf ahi = *(const v8bf*)(arow + k0 + half * 16 + 8);
    v16bf a;
#pragma unroll
    for (int i = 0; i < 8; ++i) { a[i] = alo[i]; a[i + 8] = ahi[i]; }
    const int kcol = k0 + half * 16;
#pragma unroll
    for (int j = 0; j < 4; ++j) {
      const int n = j * 16 + m16;
      v16bf b = *(const v16bf*)(W2 + (size_t)n * HID_C + kcol);
      acc[j] = __builtin_amdgcn_wmma_f32_16x16x32_bf16(
          false, a, false, b, (short)0, acc[j], false, false);
    }
  }

#pragma unroll
  for (int j = 0; j < 4; ++j) {
    const int col  = j * 16 + m16;
    const float bs = b2[col];
#pragma unroll
    for (int v = 0; v < 8; ++v) {
      const int r = mTile * 16 + v + 8 * half;
      if (r < n_nodes) h0[(size_t)r * OUT_C + col] = acc[j][v] + bs;
    }
  }
}

// ---------------- degree / dinv ----------------
__global__ void deg_init_kernel(float* __restrict__ dinv, int* __restrict__ deg_i, int n) {
  int i = blockIdx.x * 256 + threadIdx.x;
  if (i < n) { dinv[i] = 1.0f; deg_i[i] = 0; }   // dinv seeded with self-loop
}

__global__ void deg_accum_kernel(float* __restrict__ dinv, int* __restrict__ deg_i,
                                 const long long* __restrict__ dst, long long n_edges) {
  long long e = (long long)blockIdx.x * 256 + threadIdx.x;
  if (e < n_edges) {
    int d = (int)dst[e];
    atomicAdd(&dinv[d], 1.0f);   // float degree for rsqrt
    atomicAdd(&deg_i[d], 1);     // int in-degree for CSR
  }
}

__global__ void deg_rsqrt_kernel(float* __restrict__ dinv, int n) {
  int i = blockIdx.x * 256 + threadIdx.x;
  if (i < n) dinv[i] = rsqrtf(dinv[i]);
}

// ---------------- single-workgroup exclusive scan (row_ptr) ----------------
__global__ __launch_bounds__(1024) void scan_kernel(const int* __restrict__ deg_i,
                                                    int* __restrict__ row_ptr,
                                                    int* __restrict__ cursor, int n) {
  __shared__ int buf[1024];
  __shared__ int carry_s;
  const int tid = threadIdx.x;
  if (tid == 0) carry_s = 0;
  __syncthreads();
  for (int base = 0; base < n; base += 1024) {
    int i = base + tid;
    int v = (i < n) ? deg_i[i] : 0;
    buf[tid] = v;
    __syncthreads();
#pragma unroll
    for (int off = 1; off < 1024; off <<= 1) {
      int t = (tid >= off) ? buf[tid - off] : 0;
      __syncthreads();
      buf[tid] += t;
      __syncthreads();
    }
    int incl = buf[tid];
    int c = carry_s;                 // carry not modified since last barrier
    if (i < n) {
      int excl = c + incl - v;
      row_ptr[i] = excl;
      cursor[i]  = excl;
    }
    __syncthreads();
    if (tid == 1023) carry_s = c + incl;  // chunk total
    __syncthreads();
  }
  if (tid == 0) row_ptr[n] = carry_s;
}

// ---------------- CSR scatter: (src32, 0.9*norm) per slot ----------------
__global__ void csr_scatter_kernel(const long long* __restrict__ src,
                                   const long long* __restrict__ dst,
                                   const float* __restrict__ dinv,
                                   int* __restrict__ cursor,
                                   int* __restrict__ csr_src,
                                   float* __restrict__ csr_w, long long n_edges) {
  long long e = (long long)blockIdx.x * 256 + threadIdx.x;
  if (e >= n_edges) return;
  int s = (int)src[e];
  int d = (int)dst[e];
  int pos = atomicAdd(&cursor[d], 1);
  csr_src[pos] = s;
  csr_w[pos]   = 0.9f * dinv[s] * dinv[d];   // (1-alpha) folded in
}

// ---------------- propagation step (pull/gather, no atomics) ----------------
// one wave per (node, 32-channel half); lanes cover 32 consecutive channels
__global__ __launch_bounds__(256) void prop_gather_kernel(
    float* __restrict__ h_next, const float* __restrict__ h_cur,
    const float* __restrict__ h0, const float* __restrict__ dinv,
    const int* __restrict__ row_ptr, const int* __restrict__ csr_src,
    const float* __restrict__ csr_w, int n_nodes) {
  const int wgl  = blockIdx.x * 8 + (threadIdx.x >> 5);
  const int node = wgl >> 1;
  if (node >= n_nodes) return;           // wave-uniform
  const int lane = threadIdx.x & 31;
  const int c    = ((wgl & 1) << 5) + lane;

  const int beg = row_ptr[node];
  const int end = row_ptr[node + 1];
  const float di = dinv[node];
  const size_t oc = (size_t)node * OUT_C + c;

  float acc = 0.9f * di * di * h_cur[oc];  // self-loop term
  int j = beg;
  // unroll-by-2 for a little more MLP on the L2 gathers
  for (; j + 1 < end; j += 2) {
    int s0 = csr_src[j];
    int s1 = csr_src[j + 1];
    float w0 = csr_w[j];
    float w1 = csr_w[j + 1];
    acc += w0 * h_cur[(size_t)s0 * OUT_C + c];
    acc += w1 * h_cur[(size_t)s1 * OUT_C + c];
  }
  if (j < end) {
    acc += csr_w[j] * h_cur[(size_t)csr_src[j] * OUT_C + c];
  }
  h_next[oc] = acc + 0.1f * h0[oc];
}

// ---------------- launcher ----------------
extern "C" void kernel_launch(void* const* d_in, const int* in_sizes, int n_in,
                              void* d_out, int out_size, void* d_ws, size_t ws_size,
                              hipStream_t stream) {
  const float*     x    = (const float*)d_in[0];
  const float*     W1   = (const float*)d_in[1];
  const float*     b1   = (const float*)d_in[2];
  const float*     W2   = (const float*)d_in[3];
  const float*     b2   = (const float*)d_in[4];
  const long long* ei   = (const long long*)d_in[5];

  const int       n_nodes = in_sizes[0] / IN_C;
  const long long n_edges = (long long)in_sizes[5] / 2;
  const long long* src = ei;
  const long long* dst = ei + n_edges;

  // workspace carve-up (256B aligned)
  char* base = (char*)d_ws;
  size_t off = 0;
  auto carve = [&](size_t bytes) {
    void* p = base + off;
    off = (off + bytes + 255) & ~(size_t)255;
    return p;
  };
  __bf16* W1bf    = (__bf16*)carve((size_t)HID_C * IN_C * sizeof(__bf16));
  __bf16* W2bf    = (__bf16*)carve((size_t)OUT_C * HID_C * sizeof(__bf16));
  __bf16* h1      = (__bf16*)carve((size_t)n_nodes * HID_C * sizeof(__bf16));
  float*  h0      = (float*)carve((size_t)n_nodes * OUT_C * sizeof(float));
  float*  hA      = (float*)carve((size_t)n_nodes * OUT_C * sizeof(float));
  float*  hB      = (float*)carve((size_t)n_nodes * OUT_C * sizeof(float));
  float*  dinv    = (float*)carve((size_t)n_nodes * sizeof(float));
  int*    deg_i   = (int*)carve((size_t)n_nodes * sizeof(int));
  int*    row_ptr = (int*)carve((size_t)(n_nodes + 1) * sizeof(int));
  int*    cursor  = (int*)carve((size_t)n_nodes * sizeof(int));
  int*    csr_src = (int*)carve((size_t)n_edges * sizeof(int));
  float*  csr_w   = (float*)carve((size_t)n_edges * sizeof(float));

  const int mTiles = (n_nodes + 15) >> 4;

  // weights -> bf16
  cvt_f32_bf16_kernel<<<(HID_C * IN_C + 255) / 256, 256, 0, stream>>>(W1, W1bf, HID_C * IN_C);
  cvt_f32_bf16_kernel<<<(OUT_C * HID_C + 255) / 256, 256, 0, stream>>>(W2, W2bf, OUT_C * HID_C);

  // MLP via WMMA
  gemm1_kernel<<<(mTiles * 4 + 7) / 8, 256, 0, stream>>>(x, W1bf, b1, h1, n_nodes);
  gemm2_kernel<<<(mTiles + 7) / 8, 256, 0, stream>>>(h1, W2bf, b2, h0, n_nodes);

  // GCN normalization + CSR build (one-time per call)
  int nb_nodes = (n_nodes + 255) / 256;
  int nb_edges = (int)((n_edges + 255) / 256);
  deg_init_kernel<<<nb_nodes, 256, 0, stream>>>(dinv, deg_i, n_nodes);
  deg_accum_kernel<<<nb_edges, 256, 0, stream>>>(dinv, deg_i, dst, n_edges);
  deg_rsqrt_kernel<<<nb_nodes, 256, 0, stream>>>(dinv, n_nodes);
  scan_kernel<<<1, 1024, 0, stream>>>(deg_i, row_ptr, cursor, n_nodes);
  csr_scatter_kernel<<<nb_edges, 256, 0, stream>>>(src, dst, dinv, cursor,
                                                   csr_src, csr_w, n_edges);

  // APPNP propagation: 10 pull steps, ping-pong, last step writes d_out
  int nb_prop = (n_nodes * 2 + 7) / 8;   // waves = n_nodes * 2, 8 waves/block
  const float* cur = h0;
  float* bufs[2] = {hA, hB};
  for (int t = 0; t < 10; ++t) {
    float* nxt = (t == 9) ? (float*)d_out : bufs[t & 1];
    prop_gather_kernel<<<nb_prop, 256, 0, stream>>>(nxt, cur, h0, dinv,
                                                    row_ptr, csr_src, csr_w, n_nodes);
    cur = nxt;
  }
}
